// ControlledNODE_59064390255155
// MI455X (gfx1250) — matrix operand order repaired and checked by reading, not compiled, not run on hardware
//
#include <hip/hip_runtime.h>

typedef __attribute__((ext_vector_type(16))) _Float16 v16h;
typedef __attribute__((ext_vector_type(8)))  float    v8f;

#define HDIM_      32
#define UDIM_      8
#define HID_       128
#define T_         65536
#define DYN_GAIN_  0.02f
#define DAMP_      0.1f

// Build one 16x32 f16 A-operand (8 VGPRs) with the activation vector
// broadcast across all 16 rows (CDNA5 16-bit A layout).
// Per lane this is two contiguous 16-byte chunks -> 2x ds_load_b128.
__device__ __forceinline__ v16h build_A(const _Float16* __restrict__ x, int kc, int lane) {
  const int hi = (lane & 16) ? 8 : 0;
  v16h a;
#pragma unroll
  for (int v = 0; v < 8; ++v) {
    const int kb = kc * 32 + ((v & 3) << 1) + ((v >> 2) << 4) + hi;
    a[2 * v]     = x[kb];
    a[2 * v + 1] = x[kb + 1];
  }
  return a;
}

__device__ __forceinline__ v8f wmma_f16(v16h a, v16h b, v8f c) {
  return __builtin_amdgcn_wmma_f32_16x16x32_f16(false, a, false, b, (short)0, c, false, false);
}

// Fast SiLU: z * rcp(1+exp(-z))
__device__ __forceinline__ float silu_f(float z) {
  return z * __builtin_amdgcn_rcpf(1.0f + __expf(-z));
}

// Fast tanh: 1 - 2/(exp(2x)+1); exp/rcp saturate correctly at +/-inf
__device__ __forceinline__ float tanh_fast(float x) {
  return 1.0f - 2.0f * __builtin_amdgcn_rcpf(__expf(2.0f * x) + 1.0f);
}

__global__ __launch_bounds__(256) void node_scan_kernel(
    const float* __restrict__ U,   const float* __restrict__ h0,
    const float* __restrict__ W1,  const float* __restrict__ b1,
    const float* __restrict__ W2,  const float* __restrict__ b2,
    const float* __restrict__ W3,  const float* __restrict__ b3,
    const float* __restrict__ Wd,  const float* __restrict__ bd,
    const float* __restrict__ Wt,  const float* __restrict__ bt,
    const float* __restrict__ Wc,  const float* __restrict__ bc,
    float* __restrict__ out)
{
  // Per-wave private stage-input buffers: layer3 -> layer1 handoff is a
  // within-wave LDS store->load (DScnt-ordered), needing NO barrier.
  __shared__ _Float16 xbuf[8][64];     // per wave: [hh(32) | u(8) | zeros(24)]
  __shared__ _Float16 z1buf[HID_];
  __shared__ _Float16 z2buf[HID_];

  const int   tid  = threadIdx.x;
  const int   wave = tid >> 5;
  const int   lane = tid & 31;
  const int   col  = lane & 15;
  const int   n0   = wave << 4;
  const float dt   = 5.0f / 60.0f;
  const v8f   vzero = {};
  _Float16*   xb   = xbuf[wave];

  // ---------------- weights resident in VGPRs, WMMA B layout ----------------
  // B tile (32x16 f16, 8 VGPRs): lane%16 = N, VGPR v holds K = 2(v&3)+16(v>>2)+8(lane/16), +1
  v16h B1[2], B2[4], B3[2][4];
#pragma unroll
  for (int kc = 0; kc < 2; ++kc) {
    const int kb = kc * 32 + ((lane & 16) ? 8 : 0);     // K padded 40 -> 64
    v16h b;
#pragma unroll
    for (int v = 0; v < 8; ++v) {
      const int kk = kb + ((v & 3) << 1) + ((v >> 2) << 4);
      const bool ok = kk < (HDIM_ + UDIM_);
      const int n = n0 + col;
      b[2 * v]     = (_Float16)(ok ? W1[kk * HID_ + n]     : 0.0f);
      b[2 * v + 1] = (_Float16)(ok ? W1[(kk + 1) * HID_ + n] : 0.0f);
    }
    B1[kc] = b;
  }
#pragma unroll
  for (int kc = 0; kc < 4; ++kc) {
    const int kb = kc * 32 + ((lane & 16) ? 8 : 0);
    v16h b;
#pragma unroll
    for (int v = 0; v < 8; ++v) {
      const int kk = kb + ((v & 3) << 1) + ((v >> 2) << 4);
      const int n = n0 + col;
      b[2 * v]     = (_Float16)W2[kk * HID_ + n];
      b[2 * v + 1] = (_Float16)W2[(kk + 1) * HID_ + n];
    }
    B2[kc] = b;
  }
#pragma unroll
  for (int tile = 0; tile < 2; ++tile) {       // every wave holds all of W3
#pragma unroll
    for (int kc = 0; kc < 4; ++kc) {
      const int kb = kc * 32 + ((lane & 16) ? 8 : 0);
      v16h b;
#pragma unroll
      for (int v = 0; v < 8; ++v) {
        const int kk = kb + ((v & 3) << 1) + ((v >> 2) << 4);
        const int n = tile * 16 + col;
        b[2 * v]     = (_Float16)W3[kk * HDIM_ + n];
        b[2 * v + 1] = (_Float16)W3[(kk + 1) * HDIM_ + n];
      }
      B3[tile][kc] = b;
    }
  }

  const float bias1 = b1[n0 + col];
  const float bias2 = b2[n0 + col];
  const float b3c0  = b3[col];
  const float b3c1  = b3[16 + col];

  // heads: wave1 -> d, wave2 -> t, wave3 -> c (own h replica, off crit path)
  float hwv = 0.f, hbv = 0.f;
  float* outp = out;
  if (wave >= 1 && wave <= 3) {
    const float* Wh = (wave == 1) ? Wd : (wave == 2) ? Wt : Wc;
    const float* bh = (wave == 1) ? bd : (wave == 2) ? bt : bc;
    hwv  = Wh[lane];
    hbv  = bh[0];
    outp = out + (wave - 1) * T_;
  }

  // replicated per-lane state in EVERY wave (identical FP ops -> identical)
  float h  = h0[lane];
  float hh = h;

  xb[lane] = (_Float16)h;
  if (lane < UDIM_) xb[HDIM_ + lane] = (_Float16)U[lane];
  if (lane < 24)    xb[HDIM_ + UDIM_ + lane] = (_Float16)0.0f;
  __syncthreads();

  // ---------------- sequential scan: 8 barriers / step ----------------------
  for (int t = 0; t < T_; ++t) {
    float ksum = 0.f;

#pragma unroll
    for (int s = 0; s < 4; ++s) {
      // ---- layer 1: (1x64)x(64x128), from own xbuf copy -------------------
      {
        const v16h a0 = build_A(xb, 0, lane);
        const v16h a1 = build_A(xb, 1, lane);
        const v8f p0 = wmma_f16(a0, B1[0], vzero);
        const v8f p1 = wmma_f16(a1, B1[1], vzero);
        const float z = silu_f(p0[0] + p1[0] + bias1);
        if (lane < 16) z1buf[n0 + lane] = (_Float16)z;
      }
      __syncthreads();

      // ---- layer 2: (1x128)x(128x128) -------------------------------------
      {
        const v16h a0 = build_A(z1buf, 0, lane);
        const v16h a1 = build_A(z1buf, 1, lane);
        const v16h a2 = build_A(z1buf, 2, lane);
        const v16h a3 = build_A(z1buf, 3, lane);
        const v8f q0 = wmma_f16(a0, B2[0], vzero);
        const v8f q1 = wmma_f16(a1, B2[1], vzero);
        const v8f q2 = wmma_f16(a2, B2[2], vzero);
        const v8f q3 = wmma_f16(a3, B2[3], vzero);
        const float z = silu_f((q0[0] + q1[0]) + (q2[0] + q3[0]) + bias2);
        if (lane < 16) z2buf[n0 + lane] = (_Float16)z;
      }
      __syncthreads();

      // ---- layer 3 + RK bookkeeping: replicated in every wave -------------
      {
        const v16h a0 = build_A(z2buf, 0, lane);
        const v16h a1 = build_A(z2buf, 1, lane);
        const v16h a2 = build_A(z2buf, 2, lane);
        const v16h a3 = build_A(z2buf, 3, lane);
        const v8f r00 = wmma_f16(a0, B3[0][0], vzero);
        const v8f r01 = wmma_f16(a1, B3[0][1], vzero);
        const v8f r02 = wmma_f16(a2, B3[0][2], vzero);
        const v8f r03 = wmma_f16(a3, B3[0][3], vzero);
        const v8f r10 = wmma_f16(a0, B3[1][0], vzero);
        const v8f r11 = wmma_f16(a1, B3[1][1], vzero);
        const v8f r12 = wmma_f16(a2, B3[1][2], vzero);
        const v8f r13 = wmma_f16(a3, B3[1][3], vzero);
        // lane l holds drift[l%16] (tile0) and drift[16+l%16] (tile1)
        const float d0 = (r00[0] + r01[0]) + (r02[0] + r03[0]) + b3c0;
        const float d1 = (r10[0] + r11[0]) + (r12[0] + r13[0]) + b3c1;
        const float drift = (lane < 16) ? d0 : d1;
        const float kv = DYN_GAIN_ * drift - DAMP_ * hh;
        ksum += ((s == 0 || s == 3) ? 1.0f : 2.0f) * kv;

        if (s < 3) {
          const float acn = (s == 2) ? 1.0f : 0.5f;   // coeff for stage s+1
          hh = h + acn * dt * kv;
          xb[lane] = (_Float16)hh;                    // own copy, no barrier
        } else {
          float hn = h + (dt / 6.0f) * ksum;          // RK4 combine
          if (!__builtin_isfinite(hn)) hn = 0.0f;
          hn = tanh_fast(hn);
          hn = fminf(5.0f, fmaxf(-5.0f, hn));
          h = hn; hh = hn;
          xb[lane] = (_Float16)hn;                    // stage 0 input, step t+1
          if (t + 1 < T_) {
            if (lane < UDIM_)                         // own u slice (L0 hit)
              xb[HDIM_ + lane] = (_Float16)U[(t + 1) * UDIM_ + lane];
            if (wave == 7 && lane == 0 && t + 65 < T_)
              __builtin_prefetch(&U[(t + 65) * UDIM_], 0, 3);
          }
        }
      }

      if (s == 0 && wave >= 1 && wave <= 3) {
        // one head per wave on h_t (own register replica), once per step
        float v = h * hwv;
#pragma unroll
        for (int m = 16; m >= 1; m >>= 1) v += __shfl_xor(v, m);
        if (lane == 0) outp[t] = v + hbv;
      }
      // no barrier here: layer3 -> layer1 is within-wave via private xbuf;
      // z1/z2 cross-wave hazards are covered by the two barriers above.
    }
  }

  if (wave == 0) out[3 * T_ + lane] = h;   // h_last (1x32)
}

extern "C" void kernel_launch(void* const* d_in, const int* in_sizes, int n_in,
                              void* d_out, int out_size, void* d_ws, size_t ws_size,
                              hipStream_t stream) {
  (void)in_sizes; (void)n_in; (void)out_size; (void)d_ws; (void)ws_size;
  const float* U  = (const float*)d_in[0];
  const float* h0 = (const float*)d_in[1];
  const float* W1 = (const float*)d_in[2];
  const float* b1 = (const float*)d_in[3];
  const float* W2 = (const float*)d_in[4];
  const float* b2 = (const float*)d_in[5];
  const float* W3 = (const float*)d_in[6];
  const float* b3 = (const float*)d_in[7];
  const float* Wd = (const float*)d_in[8];
  const float* bd = (const float*)d_in[9];
  const float* Wt = (const float*)d_in[10];
  const float* bt = (const float*)d_in[11];
  const float* Wc = (const float*)d_in[12];
  const float* bc = (const float*)d_in[13];
  float* out = (float*)d_out;

  node_scan_kernel<<<1, 256, 0, stream>>>(U, h0, W1, b1, W2, b2, W3, b3,
                                          Wd, bd, Wt, bt, Wc, bc, out);
}
